// ResidualSTU_85349590106372
// MI455X (gfx1250) — compile-verified, matrix operand order checked by reading, not run on hardware
//
#include <hip/hip_runtime.h>
#include <hip/hip_bf16.h>
#include <math.h>

// ---------------------------------------------------------------------------
// ResidualSTU on gfx1250 (MI455X): FFT conv in LDS + bf16 WMMA for all GEMMs.
// ---------------------------------------------------------------------------

static constexpr int B     = 4;
static constexpr int SL    = 4096;
static constexpr int NPADN = 8192;
static constexpr int FFTLOG = 13;
static constexpr int Kf    = 16;
static constexpr int D     = 128;
static constexpr int KU    = 3;
static constexpr int NM    = 2;
static constexpr int NL    = 2;
static constexpr int DIN   = 64;
static constexpr int DOUT  = 64;
static constexpr int HID   = 512;
static constexpr size_t MROWS = (size_t)B * SL;           // 16384

typedef __bf16 bf16_t;
typedef bf16_t v16bf __attribute__((ext_vector_type(16)));
typedef float  v8f   __attribute__((ext_vector_type(8)));

__device__ __forceinline__ unsigned int f2bf(float f) {
  unsigned int u = __float_as_uint(f);
  u += 0x7FFFu + ((u >> 16) & 1u);      // round-to-nearest-even
  return u >> 16;
}
__device__ __forceinline__ unsigned int pack2(float a, float b) {
  return f2bf(a) | (f2bf(b) << 16);
}

// ------------------------- in-LDS 8192-pt FFT ------------------------------
__device__ __forceinline__ void fft8192(float2* buf, int tid, float sign) {
  for (int i = tid; i < NPADN; i += 256) {
    int j = (int)(__brev((unsigned)i) >> (32 - FFTLOG));
    if (j > i) { float2 t = buf[i]; buf[i] = buf[j]; buf[j] = t; }
  }
  __syncthreads();
  for (int st = 1; st <= FFTLOG; ++st) {
    const int half = 1 << (st - 1);
    const int mask = half - 1;
    const float ang = sign * 3.14159265358979323846f / (float)half;
    for (int idx = tid; idx < (NPADN >> 1); idx += 256) {
      int j    = idx & mask;
      int base = ((idx >> (st - 1)) << st);
      float s, c;
      __sincosf(ang * (float)j, &s, &c);
      float2 u = buf[base + j];
      float2 v = buf[base + j + half];
      float vr = v.x * c - v.y * s;
      float vi = v.x * s + v.y * c;
      buf[base + j]        = make_float2(u.x + vr, u.y + vi);
      buf[base + j + half] = make_float2(u.x - vr, u.y - vi);
    }
    __syncthreads();
  }
}

// Vfull[f*16+k] = FFT_8192(phi[:,k] * sigma[k]^0.25)
__global__ __launch_bounds__(256)
void stu_prep_filters(const float* __restrict__ phi, const float* __restrict__ sigma,
                      float2* __restrict__ Vfull) {
  __shared__ float2 buf[NPADN];
  const int k = blockIdx.x, tid = threadIdx.x;
  const float sc = powf(sigma[k], 0.25f);
  for (int s = tid; s < NPADN; s += 256)
    buf[s] = (s < SL) ? make_float2(phi[(size_t)s * Kf + k] * sc, 0.f)
                      : make_float2(0.f, 0.f);
  __syncthreads();
  fft8192(buf, tid, -1.f);
  for (int f = tid; f < NPADN; f += 256) Vfull[(size_t)f * Kf + k] = buf[f];
}

// Full-spectrum forward FFT of one (b, channel i) signal.
__global__ __launch_bounds__(256)
void stu_fft_fwd(const float* __restrict__ XR, float2* __restrict__ S) {
  __shared__ float2 buf[NPADN];
  const int blk = blockIdx.x, tid = threadIdx.x;
  const int b = blk >> 7, i = blk & 127;
  const float* src = XR + (size_t)b * SL * D + i;
  for (int s = tid; s < NPADN; s += 256)
    buf[s] = (s < SL) ? make_float2(src[(size_t)s * D], 0.f) : make_float2(0.f, 0.f);
  __syncthreads();
  fft8192(buf, tid, -1.f);
  float2* out = S + ((size_t)(b * D + i)) * NPADN;
  for (int f = tid; f < NPADN; f += 256) out[f] = buf[f];
}

// Multiply spectrum by filter (minus branch = half-spectrum shift), iFFT,
// store first SL real samples as bf16 into UU[(b,s), sgn*2048 + k*128 + i].
__global__ __launch_bounds__(256)
void stu_spectral_ifft(const float2* __restrict__ S, const float2* __restrict__ Vfull,
                       unsigned short* __restrict__ UU) {
  __shared__ float2 buf[NPADN];
  const int blk = blockIdx.x, tid = threadIdx.x;
  const int i = blk & 127, k = (blk >> 7) & 15, sgn = (blk >> 11) & 1, b = blk >> 12;
  const float2* sp = S + ((size_t)(b * D + i)) * NPADN;
  for (int f = tid; f < NPADN; f += 256) {
    int fs = sgn ? ((f + (NPADN >> 1)) & (NPADN - 1)) : f;
    float2 u = sp[fs];
    float2 v = Vfull[(size_t)f * Kf + k];
    buf[f] = make_float2(u.x * v.x - u.y * v.y, u.x * v.y + u.y * v.x);
  }
  __syncthreads();
  fft8192(buf, tid, +1.f);
  const int col = (sgn << 11) + (k << 7) + i;
  const float inv = 1.0f / (float)NPADN;
  unsigned short* dst = UU + (size_t)b * SL * 4096 + col;
  for (int s = tid; s < SL; s += 256)
    dst[(size_t)s * 4096] = (unsigned short)f2bf(buf[s].x * inv);
}

// ---------------------- WMMA bf16 GEMM: C = A @ W^T ------------------------
// A: (M,K) bf16 row-major.  W: (N,K) bf16 row-major.  C: (M,N) f32.
// Block tile 128x128, 8 waves = 4(M-strip of 32) x 2(N-strip of 64); each wave
// computes 2x4 = 8 16x16 WMMA tiles per K-step (K stepped by 32).
// Staging is 16B-vectorized with register double-buffering so the next K-tile's
// global_load_b128s retire underneath the current tile's 8 WMMAs.
__global__ __launch_bounds__(256)
void gemm_bf16(const unsigned short* __restrict__ A, int lda,
               const unsigned short* __restrict__ W, int ldw,
               float* __restrict__ C, int ldc,
               int M, int N, int K, int accumulate) {
  __shared__ unsigned short At[128][32];   // 8 KB
  __shared__ unsigned short Wt[128][32];   // 8 KB
  const int tid  = threadIdx.x;
  const int lane = tid & 31;
  const int wave = tid >> 5;
  const int wm = wave >> 1;                // 0..3 : 32-row M strip
  const int wn = wave & 1;                 // 0..1 : 64-col N strip
  const int bm = blockIdx.x * 128, bn = blockIdx.y * 128;
  const int row  = lane & 15;
  const int half = lane >> 4;

  // staging map: each thread moves 4 x 16B chunks (2 A rows, 2 W rows)
  const int r0 = tid >> 2;          // 0..63
  const int q  = tid & 3;           // 16B chunk within 64B row
  int wr0 = bn + r0;       if (wr0 >= N) wr0 = N - 1;   // clamp (N=64 case)
  int wr1 = bn + r0 + 64;  if (wr1 >= N) wr1 = N - 1;
  const uint4* a0 = (const uint4*)(A + (size_t)(bm + r0) * lda) + q;
  const uint4* a1 = (const uint4*)(A + (size_t)(bm + r0 + 64) * lda) + q;
  const uint4* w0 = (const uint4*)(W + (size_t)wr0 * ldw) + q;
  const uint4* w1 = (const uint4*)(W + (size_t)wr1 * ldw) + q;
  uint4* sA0 = (uint4*)&At[r0][0] + q;
  uint4* sA1 = (uint4*)&At[r0 + 64][0] + q;
  uint4* sW0 = (uint4*)&Wt[r0][0] + q;
  uint4* sW1 = (uint4*)&Wt[r0 + 64][0] + q;

  v8f acc[2][4];
#pragma unroll
  for (int a = 0; a < 2; ++a)
#pragma unroll
    for (int w = 0; w < 4; ++w) acc[a][w] = v8f{0.f,0.f,0.f,0.f,0.f,0.f,0.f,0.f};

  uint4 ra0 = a0[0], ra1 = a1[0], rw0 = w0[0], rw1 = w1[0];

  for (int kb = 0; kb < K; kb += 32) {
    *sA0 = ra0; *sA1 = ra1; *sW0 = rw0; *sW1 = rw1;
    __syncthreads();
    if (kb + 32 < K) {                    // double-buffer: fetch next K-tile
      const int kq = (kb + 32) >> 3;
      ra0 = a0[kq]; ra1 = a1[kq]; rw0 = w0[kq]; rw1 = w1[kq];
      __builtin_prefetch(a0 + kq + 4, 0, 3);   // CDNA5 global_prefetch_b8
      __builtin_prefetch(a1 + kq + 4, 0, 3);
    }
    // per-lane fragments (CDNA5 §7.12.2): two contiguous 8-element runs
    v16bf af[2], wf[4];
#pragma unroll
    for (int a = 0; a < 2; ++a) {
      const bf16_t* p = (const bf16_t*)&At[wm * 32 + a * 16 + row][half * 8];
#pragma unroll
      for (int e = 0; e < 8; ++e) { af[a][e] = p[e]; af[a][8 + e] = p[16 + e]; }
    }
#pragma unroll
    for (int w = 0; w < 4; ++w) {
      const bf16_t* p = (const bf16_t*)&Wt[wn * 64 + w * 16 + row][half * 8];
#pragma unroll
      for (int e = 0; e < 8; ++e) { wf[w][e] = p[e]; wf[w][8 + e] = p[16 + e]; }
    }
#pragma unroll
    for (int a = 0; a < 2; ++a)
#pragma unroll
      for (int w = 0; w < 4; ++w)
        acc[a][w] = __builtin_amdgcn_wmma_f32_16x16x32_bf16(
            false, af[a], false, wf[w], (short)0, acc[a][w], false, false);
    __syncthreads();
  }

#pragma unroll
  for (int a = 0; a < 2; ++a) {
#pragma unroll
    for (int w = 0; w < 4; ++w) {
      const int n = bn + wn * 64 + w * 16 + row;
      if (n < N) {
#pragma unroll
        for (int r = 0; r < 8; ++r) {
          const int m = bm + wm * 32 + a * 16 + r + half * 8;
          size_t o = (size_t)m * ldc + n;
          C[o] = accumulate ? (C[o] + acc[a][w][r]) : acc[a][w][r];
        }
      }
    }
  }
}

// ------------------------------ small kernels ------------------------------
// vec4 over feature dim: count4 = rows * D/4
__global__ void stu_embed(const int* __restrict__ inp, const float* __restrict__ Wemb,
                          float* __restrict__ X, size_t count4) {
  size_t idx = (size_t)blockIdx.x * blockDim.x + threadIdx.x;
  if (idx >= count4) return;
  size_t r = idx >> 5; int d4 = (int)(idx & 31);          // D/4 = 32
  ((float4*)X)[idx] = ((const float4*)(Wemb + (size_t)inp[r] * D))[d4];
}

__global__ __launch_bounds__(128)
void stu_rmsnorm(const float* __restrict__ X, const float* __restrict__ w,
                 float* __restrict__ XR, unsigned short* __restrict__ XRb) {
  __shared__ float red[128];
  const int r = blockIdx.x, d = threadIdx.x;
  float v = X[(size_t)r * D + d];
  red[d] = v * v;
  __syncthreads();
  for (int s = 64; s > 0; s >>= 1) { if (d < s) red[d] += red[d + s]; __syncthreads(); }
  float o = v * rsqrtf(red[0] * (1.0f / D) + 1e-6f) * w[d];
  XR[(size_t)r * D + d]  = o;
  XRb[(size_t)r * D + d] = (unsigned short)f2bf(o);
}

// AA[(r), i*128+d] = XRb[r - i][d] (0 if shifted past sequence start); vec4.
__global__ void stu_build_arA(const unsigned short* __restrict__ XRb,
                              unsigned short* __restrict__ AA, size_t count4) {
  size_t idx = (size_t)blockIdx.x * blockDim.x + threadIdx.x;
  if (idx >= count4) return;
  size_t r = idx / (KU * 32); int kk4 = (int)(idx - r * (KU * 32));
  int i = kk4 >> 5, d4 = kk4 & 31;
  int s = (int)(r & (SL - 1));
  uint2 v = make_uint2(0u, 0u);
  if (s >= i) v = ((const uint2*)(XRb + (r - i) * D))[d4];
  ((uint2*)AA)[idx] = v;
}

__global__ void stu_add_cast(float* __restrict__ X, const float* __restrict__ Y,
                             unsigned short* __restrict__ Xb, size_t count4) {
  size_t idx = (size_t)blockIdx.x * blockDim.x + threadIdx.x;
  if (idx >= count4) return;
  float4 x = ((float4*)X)[idx];
  float4 y = ((const float4*)Y)[idx];
  x.x += y.x; x.y += y.y; x.z += y.z; x.w += y.w;
  ((float4*)X)[idx] = x;
  ((uint2*)Xb)[idx] = make_uint2(pack2(x.x, x.y), pack2(x.z, x.w));
}

__global__ void stu_add3(float* __restrict__ X, const float* __restrict__ Y,
                         const float* __restrict__ Z, size_t count4) {
  size_t idx = (size_t)blockIdx.x * blockDim.x + threadIdx.x;
  if (idx >= count4) return;
  float4 x = ((float4*)X)[idx];
  float4 y = ((const float4*)Y)[idx];
  float4 z = ((const float4*)Z)[idx];
  x.x += y.x + z.x; x.y += y.y + z.y; x.z += y.z + z.z; x.w += y.w + z.w;
  ((float4*)X)[idx] = x;
}

__global__ void stu_copy_f32(const float* __restrict__ S, float* __restrict__ Dst,
                             size_t count4) {
  size_t idx = (size_t)blockIdx.x * blockDim.x + threadIdx.x;
  if (idx < count4) ((float4*)Dst)[idx] = ((const float4*)S)[idx];
}

__global__ void stu_cast_bf16(const float* __restrict__ S, unsigned short* __restrict__ Dst,
                              size_t count4) {
  size_t idx = (size_t)blockIdx.x * blockDim.x + threadIdx.x;
  if (idx >= count4) return;
  float4 v = ((const float4*)S)[idx];
  ((uint2*)Dst)[idx] = make_uint2(pack2(v.x, v.y), pack2(v.z, v.w));
}

__global__ void stu_swiglu(const float* __restrict__ H2, unsigned short* __restrict__ Hb,
                           size_t count4) {   // count4 = MROWS * HID/4
  size_t idx = (size_t)blockIdx.x * blockDim.x + threadIdx.x;
  if (idx >= count4) return;
  size_t r = idx >> 7; int j4 = (int)(idx & 127);          // HID/4 = 128
  const float4* Hrow = (const float4*)(H2 + r * (2 * HID));
  float4 y = Hrow[j4];
  float4 g = Hrow[128 + j4];
  y.x *= g.x / (1.f + __expf(-g.x));
  y.y *= g.y / (1.f + __expf(-g.y));
  y.z *= g.z / (1.f + __expf(-g.z));
  y.w *= g.w / (1.f + __expf(-g.w));
  ((uint2*)Hb)[idx] = make_uint2(pack2(y.x, y.y), pack2(y.z, y.w));
}

// pack M_phi_plus/minus into Wpm[(ml, o), k*128+i | 2048+...] bf16
__global__ void stu_pack_wpm(const float* __restrict__ Mp, const float* __restrict__ Mm,
                             unsigned short* __restrict__ Wpm, size_t count) {
  size_t idx = (size_t)blockIdx.x * blockDim.x + threadIdx.x;
  if (idx >= count) return;
  size_t ro = idx >> 12; int col = (int)(idx & 4095);
  int o = (int)(ro & 127); size_t ml = ro >> 7;
  const float* Msrc = (col < 2048) ? Mp : Mm;
  int c = col & 2047;
  int k = c >> 7, i = c & 127;
  Wpm[idx] = (unsigned short)f2bf(Msrc[((ml * Kf + k) * D + i) * D + o]);
}

// pack M_u into Wu[(ml, o), i*128+d] bf16
__global__ void stu_pack_wu(const float* __restrict__ Mu, unsigned short* __restrict__ Wu,
                            size_t count) {
  size_t idx = (size_t)blockIdx.x * blockDim.x + threadIdx.x;
  if (idx >= count) return;
  size_t ro = idx / (KU * D); int col = (int)(idx - ro * (KU * D));
  int o = (int)(ro & 127); size_t ml = ro >> 7;
  int i = col >> 7, d = col & 127;
  Wu[idx] = (unsigned short)f2bf(Mu[((ml * KU + i) * D + d) * D + o]);
}

// ---------------------------------------------------------------------------
extern "C" void kernel_launch(void* const* d_in, const int* in_sizes, int n_in,
                              void* d_out, int out_size, void* d_ws, size_t ws_size,
                              hipStream_t stream) {
  const int*   inputs  = (const int*)  d_in[0];
  // d_in[1] = targets: unused (residual never returned)
  const float* sigma   = (const float*)d_in[2];
  const float* phi     = (const float*)d_in[3];
  const float* embed_w = (const float*)d_in[4];
  const float* rn_w    = (const float*)d_in[5];
  const float* M_u     = (const float*)d_in[6];
  const float* M_pp    = (const float*)d_in[7];
  const float* M_pm    = (const float*)d_in[8];
  const float* fc1_w   = (const float*)d_in[9];
  const float* fc2_w   = (const float*)d_in[10];
  const float* out_w   = (const float*)d_in[11];
  float* OUT = (float*)d_out;

  char* ws = (char*)d_ws;
  size_t off = 0;
  auto alloc = [&](size_t bytes) -> char* {
    char* p = ws + off;
    off += (bytes + 255) & ~(size_t)255;
    return p;
  };
  float2*         Vfull = (float2*)        alloc((size_t)NPADN * Kf * sizeof(float2));
  float2*         S     = (float2*)        alloc((size_t)B * D * NPADN * sizeof(float2));
  unsigned short* UU    = (unsigned short*)alloc(MROWS * 4096 * 2);
  float*          X     = (float*)         alloc(MROWS * D * 4);
  float*          Z     = (float*)         alloc(MROWS * D * 4);
  float*          XR    = (float*)         alloc(MROWS * D * 4);
  unsigned short* XRb   = (unsigned short*)alloc(MROWS * D * 2);
  unsigned short* Xb    = (unsigned short*)alloc(MROWS * D * 2);
  float*          Y     = (float*)         alloc(MROWS * D * 4);
  unsigned short* AA    = (unsigned short*)alloc(MROWS * KU * D * 2);
  float*          H2    = (float*)         alloc(MROWS * 2 * HID * 4);
  unsigned short* Hb    = (unsigned short*)alloc(MROWS * HID * 2);
  unsigned short* Wpm   = (unsigned short*)alloc((size_t)NM * NL * D * 4096 * 2);
  unsigned short* Wu    = (unsigned short*)alloc((size_t)NM * NL * D * KU * D * 2);
  unsigned short* Wfc1  = (unsigned short*)alloc((size_t)NM * NL * 2 * HID * D * 2);
  unsigned short* Wfc2  = (unsigned short*)alloc((size_t)NM * NL * D * HID * 2);
  unsigned short* Wout  = (unsigned short*)alloc((size_t)NM * DOUT * D * 2);
  (void)ws_size; (void)in_sizes; (void)n_in; (void)out_size;

  auto blocks = [](size_t c) { return dim3((unsigned)((c + 255) / 256)); };
  auto ggrid  = [](size_t M, int N) {
    return dim3((unsigned)(M / 128), (unsigned)((N + 127) / 128));
  };

  // ---- prep (deterministic every call) ----
  {
    size_t c1 = (size_t)NM * NL * D * 4096;
    stu_pack_wpm<<<blocks(c1), 256, 0, stream>>>(M_pp, M_pm, Wpm, c1);
    size_t c2 = (size_t)NM * NL * D * KU * D;
    stu_pack_wu<<<blocks(c2), 256, 0, stream>>>(M_u, Wu, c2);
    size_t c3 = (size_t)NM * NL * 2 * HID * D / 4;
    stu_cast_bf16<<<blocks(c3), 256, 0, stream>>>(fc1_w, Wfc1, c3);
    size_t c4 = (size_t)NM * NL * D * HID / 4;
    stu_cast_bf16<<<blocks(c4), 256, 0, stream>>>(fc2_w, Wfc2, c4);
    size_t c5 = (size_t)NM * DOUT * D / 4;
    stu_cast_bf16<<<blocks(c5), 256, 0, stream>>>(out_w, Wout, c5);
    stu_prep_filters<<<dim3(Kf), 256, 0, stream>>>(phi, sigma, Vfull);
  }

  const size_t cXD4 = MROWS * D / 4;

  for (int m = 0; m < NM; ++m) {
    stu_embed<<<blocks(cXD4), 256, 0, stream>>>(inputs, embed_w + (size_t)m * DIN * D,
                                                X, cXD4);
    for (int l = 0; l < NL; ++l) {
      const size_t ml = (size_t)m * NL + l;
      stu_copy_f32<<<blocks(cXD4), 256, 0, stream>>>(X, Z, cXD4);
      stu_rmsnorm<<<dim3((unsigned)MROWS), 128, 0, stream>>>(X, rn_w + ml * D, XR, XRb);

      // spectral conv: fwd FFT (512 wg), filter + iFFT (16384 wg)
      stu_fft_fwd<<<dim3(B * D), 256, 0, stream>>>(XR, S);
      stu_spectral_ifft<<<dim3(B * 2 * Kf * D), 256, 0, stream>>>(S, Vfull, UU);

      // Y = UU @ Wpm^T   (M=16384, N=128, K=4096): UU streamed from HBM once
      gemm_bf16<<<ggrid(MROWS, D), 256, 0, stream>>>(
          UU, 4096, Wpm + ml * D * 4096, 4096, Y, D, (int)MROWS, D, 4096, 0);

      // AR: Y += [x, x>>1, x>>2] @ Wu^T   (K=384)
      size_t cAA4 = MROWS * KU * D / 4;
      stu_build_arA<<<blocks(cAA4), 256, 0, stream>>>(XRb, AA, cAA4);
      gemm_bf16<<<ggrid(MROWS, D), 256, 0, stream>>>(
          AA, KU * D, Wu + ml * D * KU * D, KU * D, Y, D, (int)MROWS, D, KU * D, 1);

      // x = x + stu(x);  Xb = bf16(x)
      stu_add_cast<<<blocks(cXD4), 256, 0, stream>>>(X, Y, Xb, cXD4);

      // MLP: H2 = Xb @ fc1^T ; Hb = y*silu(gate) ; Y = Hb @ fc2^T
      gemm_bf16<<<ggrid(MROWS, 2 * HID), 256, 0, stream>>>(
          Xb, D, Wfc1 + ml * 2 * HID * D, D, H2, 2 * HID, (int)MROWS, 2 * HID, D, 0);
      size_t cH4 = MROWS * HID / 4;
      stu_swiglu<<<blocks(cH4), 256, 0, stream>>>(H2, Hb, cH4);
      gemm_bf16<<<ggrid(MROWS, D), 256, 0, stream>>>(
          Hb, HID, Wfc2 + ml * D * HID, HID, Y, D, (int)MROWS, D, HID, 0);

      // x = x + mlp + z
      stu_add3<<<blocks(cXD4), 256, 0, stream>>>(X, Y, Z, cXD4);
    }
    // logits accumulation: OUT (+)= X @ out_w[m]^T
    stu_cast_bf16<<<blocks(cXD4), 256, 0, stream>>>(X, Xb, cXD4);
    gemm_bf16<<<ggrid(MROWS, DOUT), 256, 0, stream>>>(
        Xb, D, Wout + (size_t)m * DOUT * D, D, OUT, DOUT, (int)MROWS, DOUT, D, m ? 1 : 0);
  }
}
